// ProbSparseSelfAttention_22497038696945
// MI455X (gfx1250) — matrix operand order repaired and checked by reading, not compiled, not run on hardware
//
#include <hip/hip_runtime.h>
#include <hip/hip_bf16.h>
#include <cmath>

typedef unsigned short u16;
typedef unsigned int   u32;

typedef __attribute__((ext_vector_type(16))) __bf16 v16bf;
typedef __attribute__((ext_vector_type(8)))  float  v8f;

struct __align__(16) U128 { u32 w[4]; };

#define EMBED  1024
#define SEQ    4096
#define BATCH  8
#define HEADS  16
#define HDIM   64
#define KQ     64          // selected queries padded to a full 64-row tile
#define CHUNKS 16

// ---------------------------------------------------------------- helpers
__device__ __forceinline__ u16 f2bf(float f) {
  u32 u = __float_as_uint(f);
  u32 r = (u + 0x7FFFu + ((u >> 16) & 1u)) >> 16;   // round-to-nearest-even
  return (u16)r;
}

// Assemble a 16-element bf16 WMMA fragment from two 16-byte LDS chunks.
__device__ __forceinline__ v16bf load_frag16(const u16* p0, const u16* p1) {
  U128 lo = *(const U128*)p0;
  U128 hi = *(const U128*)p1;
  v16bf f;
  __builtin_memcpy(&f, &lo, 16);
  __builtin_memcpy(reinterpret_cast<char*>(&f) + 16, &hi, 16);
  return f;
}

// CDNA5 async global->LDS copy, 16B per lane, tracked by ASYNCcnt.
__device__ __forceinline__ void async_ld_b128(u32 lds_addr, const u16* gptr) {
  asm volatile("global_load_async_to_lds_b128 %0, %1, off"
               :: "v"(lds_addr), "v"(gptr)
               : "memory");
}
template<int N>
__device__ __forceinline__ void wait_async() {
  asm volatile("s_wait_asynccnt %0" :: "i"(N) : "memory");
}

// ---------------------------------------------------------------- WMMA GEMM
// C = alpha * A[M,K] * op(B) + bias;  A,B bf16, C fp32 or bf16.
// B_NK = true : B stored row-major [N,K] (A @ W^T); async-staged.
// B_NK = false: B stored row-major [K,N] (A @ B);  register transpose staging.
// Double-buffered LDS software pipeline: loads for tile k+1 are in flight
// (ASYNCcnt) while tile k is computed.  All tiles are full (launches
// guarantee exact multiples of MT/NT/KT).
// Batched over gridDim.z with z = b*Hn + h; per-operand offsets b*bs + h*hs.
template<int MT, int NT, int WM, int WN, bool B_NK, bool OUT_BF>
__global__ __launch_bounds__(256) void gemm_kernel(
    const u16* __restrict__ A, const u16* __restrict__ Bm, void* __restrict__ Cv,
    const float* __restrict__ bias, float alpha,
    int K, int lda, int ldb, int ldc,
    int Hn, long bsA, long hsA, long bsB, long hsB, long bsC, long hsC)
{
  constexpr int KT = 32;
  constexpr int SM = MT / WM, SN = NT / WN;
  constexpr int TM = SM / 16, TN = SN / 16;
  constexpr int ACH = (MT * KT) / (256 * 8);   // 16B chunks per thread (A)
  constexpr int BCH = (NT * KT) / (256 * 8);   // 16B chunks per thread (B)
  constexpr int INFLIGHT = ACH + (B_NK ? BCH : 0);
  static_assert(WM * WN == 8 && TM >= 1 && TN >= 1, "8 waves, >=1 tile each");
  static_assert((MT * KT) % (256 * 8) == 0, "A tile = whole 16B chunks/thread");
  static_assert((NT * KT) % (256 * 8) == 0, "B tile = whole 16B chunks/thread");
  __shared__ __align__(16) u16 As[2 * MT * KT];
  __shared__ __align__(16) u16 Bs[2 * NT * KT];

  const int z  = blockIdx.z;
  const int bb = z / Hn, hh = z % Hn;
  A  += bb * bsA + hh * hsA;
  Bm += bb * bsB + hh * hsB;
  const long coff = bb * bsC + hh * hsC;

  const int m0 = blockIdx.y * MT;
  const int n0 = blockIdx.x * NT;
  const int t    = threadIdx.x;
  const int wave = t >> 5, lane = t & 31;
  const int wm = wave / WN, wn = wave % WN;
  const int lrow = lane & 15;        // A row / B col within 16x16 tile
  const int hf   = lane >> 4;        // lane half
  const int akb  = hf * 8;           // A K-base (ISA 16-bit A layout)
  const int bkb  = hf * 16;          // unused; B fragments use akb, same k order as A

  const u32 asb = (u32)(uintptr_t)&As[0];
  const u32 bsb = (u32)(uintptr_t)&Bs[0];

  // Stage one K-tile into LDS buffer `buf`.
  auto stage = [&](int buf, int k0) {
    const u32 abase = asb + (u32)buf * (MT * KT * 2);
#pragma unroll
    for (int i = 0; i < ACH; ++i) {
      int ch = t + i * 256;
      int r = ch >> 2, c = (ch & 3) * 8;      // 4 chunks per 32-elem row
      async_ld_b128(abase + ch * 16, A + (long)(m0 + r) * lda + k0 + c);
    }
    const u32 bbase = bsb + (u32)buf * (NT * KT * 2);
    if constexpr (B_NK) {
#pragma unroll
      for (int i = 0; i < BCH; ++i) {
        int ch = t + i * 256;
        int r = ch >> 2, c = (ch & 3) * 8;
        async_ld_b128(bbase + ch * 16, Bm + (long)(n0 + r) * ldb + k0 + c);
      }
    } else {
      // B is [K,N]: registers + transposed LDS stores (full tile, no bounds)
      u16* bsl = &Bs[buf * (NT * KT)];
#pragma unroll
      for (int i = 0; i < BCH; ++i) {
        int ch = t + i * 256;
        int kk2 = ch / (NT / 8);
        int nc  = (ch % (NT / 8)) * 8;
        U128 v = *(const U128*)(Bm + (long)(k0 + kk2) * ldb + n0 + nc);
        const u16* pv = (const u16*)&v;
#pragma unroll
        for (int q = 0; q < 8; ++q)
          bsl[(nc + q) * KT + kk2] = pv[q];
      }
    }
  };

  v8f acc[TM][TN];
  for (int i = 0; i < TM; ++i)
    for (int j = 0; j < TN; ++j)
      acc[i][j] = v8f{0.f,0.f,0.f,0.f,0.f,0.f,0.f,0.f};

  stage(0, 0);                                // prologue: tile 0 in flight
  int cur = 0;
  for (int k0 = 0; k0 < K; k0 += KT) {
    const bool more = (k0 + KT) < K;
    if (more) stage(cur ^ 1, k0 + KT);        // next tile in flight
    // prefetch two tiles ahead into cache
    int nk2 = k0 + 2 * KT;
    if (nk2 < K) {
      if (t < MT) __builtin_prefetch(A + (long)(m0 + t) * lda + nk2, 0, 3);
      else if (B_NK && (t - MT) < NT)
        __builtin_prefetch(Bm + (long)(n0 + t - MT) * ldb + nk2, 0, 3);
    }
    if (more) wait_async<INFLIGHT>();         // only newest loads outstanding
    else      wait_async<0>();
    __syncthreads();

    // ---- fragments per ISA 7.12.2 layouts, then WMMA
    const u16* Ab = &As[cur * (MT * KT)];
    const u16* Bb = &Bs[cur * (NT * KT)];
    v16bf af[TM], bfr[TN];
#pragma unroll
    for (int i = 0; i < TM; ++i) {
      const u16* base = &Ab[(wm * SM + i * 16 + lrow) * KT];
      af[i] = load_frag16(base + akb, base + 16 + akb);
    }
#pragma unroll
    for (int j = 0; j < TN; ++j) {
      const u16* base = &Bb[(wn * SN + j * 16 + lrow) * KT];
      bfr[j] = load_frag16(base + akb, base + 16 + akb);
    }
#pragma unroll
    for (int i = 0; i < TM; ++i)
#pragma unroll
      for (int j = 0; j < TN; ++j)
        acc[i][j] = __builtin_amdgcn_wmma_f32_16x16x32_bf16(
            false, af[i], false, bfr[j], (short)0, acc[i][j], false, false);
    __syncthreads();
    cur ^= 1;
  }

  // ---- epilogue: C/D layout lane = col, VGPR r -> row r + 8*half (full tiles)
  float* Cf = (float*)Cv;
  u16*   Cb = (u16*)Cv;
#pragma unroll
  for (int i = 0; i < TM; ++i) {
    int gmb = m0 + wm * SM + i * 16 + hf * 8;
#pragma unroll
    for (int j = 0; j < TN; ++j) {
      int gn = n0 + wn * SN + j * 16 + lrow;
      float badd = bias ? bias[gn] : 0.f;
#pragma unroll
      for (int r = 0; r < 8; ++r) {
        float v = acc[i][j][r] * alpha + badd;
        long off = coff + (long)(gmb + r) * ldc + gn;
        if (OUT_BF) Cb[off] = f2bf(v);
        else        Cf[off] = v;
      }
    }
  }
}

// ---------------------------------------------------------------- misc kernels
__global__ void cvt_bf16_kernel(const float* __restrict__ in, u16* __restrict__ out, long n) {
  long i = blockIdx.x * 256L + threadIdx.x;
  long stride = gridDim.x * 256L;
  for (; i < n; i += stride) out[i] = f2bf(in[i]);
}

__global__ __launch_bounds__(256) void energy_kernel(const float* __restrict__ x,
                                                     float* __restrict__ energy) {
  __shared__ float red[256];
  long tok = blockIdx.x;                 // B*S tokens
  const float* r = x + tok * EMBED;
  float s = 0.f;
  for (int i = threadIdx.x; i < EMBED; i += 256) { float v = r[i]; s += v * v; }
  red[threadIdx.x] = s; __syncthreads();
  for (int o = 128; o > 0; o >>= 1) {
    if (threadIdx.x < o) red[threadIdx.x] += red[threadIdx.x + o];
    __syncthreads();
  }
  if (!threadIdx.x) energy[tok] = sqrtf(red[0]);
}

__global__ __launch_bounds__(256) void colsum_kernel(const float* __restrict__ x,
                                                     float* __restrict__ partial) {
  int e = blockIdx.x * 256 + threadIdx.x;
  int ch = blockIdx.y, b = blockIdx.z;
  int s0 = ch * (SEQ / CHUNKS);
  const float* p = x + ((long)b * SEQ + s0) * EMBED + e;
  float s = 0.f;
  for (int i = 0; i < SEQ / CHUNKS; ++i) s += p[(long)i * EMBED];
  partial[(long)ch * (BATCH * EMBED) + b * EMBED + e] = s;
}

__global__ void colreduce_kernel(const float* __restrict__ partial, float* __restrict__ defs) {
  int i = blockIdx.x * 256 + threadIdx.x;            // over B*E
  float s = 0.f;
  for (int ch = 0; ch < CHUNKS; ++ch) s += partial[(long)ch * (BATCH * EMBED) + i];
  defs[i] = s * (1.0f / SEQ);
}

__global__ __launch_bounds__(256) void topk_kernel(const float* __restrict__ energy,
                                                   int* __restrict__ topk, int k) {
  __shared__ float sv[256];
  __shared__ int   si[256];
  __shared__ u32   sel[SEQ / 32];
  int b = blockIdx.x;
  const float* e = energy + (long)b * SEQ;
  for (int i = threadIdx.x; i < SEQ / 32; i += 256) sel[i] = 0;
  __syncthreads();
  for (int it = 0; it < k; ++it) {
    float bv = -3.0e38f; int bi = 0x7FFFFFFF;
    for (int s = threadIdx.x; s < SEQ; s += 256) {
      if ((sel[s >> 5] >> (s & 31)) & 1u) continue;
      float v = e[s];
      if (v > bv || (v == bv && s < bi)) { bv = v; bi = s; }
    }
    sv[threadIdx.x] = bv; si[threadIdx.x] = bi; __syncthreads();
    for (int o = 128; o > 0; o >>= 1) {
      if (threadIdx.x < o) {
        float v2 = sv[threadIdx.x + o]; int i2 = si[threadIdx.x + o];
        if (v2 > sv[threadIdx.x] || (v2 == sv[threadIdx.x] && i2 < si[threadIdx.x])) {
          sv[threadIdx.x] = v2; si[threadIdx.x] = i2;
        }
      }
      __syncthreads();
    }
    if (!threadIdx.x) { topk[b * 64 + it] = si[0]; sel[si[0] >> 5] |= 1u << (si[0] & 31); }
    __syncthreads();
  }
}

__global__ __launch_bounds__(256) void gather_q_kernel(const float* __restrict__ x,
                                                       const int* __restrict__ topk,
                                                       u16* __restrict__ qtok, int k) {
  int rowi = blockIdx.x;                 // B*KQ rows
  int b = rowi / KQ, i = rowi % KQ;
  u16* dst = qtok + (long)rowi * EMBED;
  if (i < k) {
    const float* src = x + ((long)b * SEQ + topk[b * 64 + i]) * EMBED;
    for (int e = threadIdx.x; e < EMBED; e += 256) dst[e] = f2bf(src[e]);
  } else {
    for (int e = threadIdx.x; e < EMBED; e += 256) dst[e] = 0;
  }
}

__global__ __launch_bounds__(256) void softmax_kernel(const float* __restrict__ scores,
                                                      u16* __restrict__ attn) {
  __shared__ float rowbuf[SEQ];
  __shared__ float red[256];
  long row = blockIdx.x;                 // B*H*KQ rows
  const float* src = scores + row * (long)SEQ;
  u16* dst = attn + row * (long)SEQ;
  int t = threadIdx.x;
  float m = -3.0e38f;
  for (int i = t; i < SEQ; i += 256) { float v = src[i]; rowbuf[i] = v; m = fmaxf(m, v); }
  red[t] = m; __syncthreads();
  for (int o = 128; o > 0; o >>= 1) {
    if (t < o) red[t] = fmaxf(red[t], red[t + o]);
    __syncthreads();
  }
  float rmax = red[0]; __syncthreads();
  float s = 0.f;
  for (int i = t; i < SEQ; i += 256) { float e2 = __expf(rowbuf[i] - rmax); rowbuf[i] = e2; s += e2; }
  red[t] = s; __syncthreads();
  for (int o = 128; o > 0; o >>= 1) {
    if (t < o) red[t] += red[t + o];
    __syncthreads();
  }
  float inv = 1.f / red[0];
  for (int i = t; i < SEQ; i += 256) dst[i] = f2bf(rowbuf[i] * inv);
}

__global__ void fill_kernel(const float* __restrict__ defs, float* __restrict__ out) {
  long n = (long)BATCH * SEQ * EMBED;
  long i = blockIdx.x * 256L + threadIdx.x;
  long stride = gridDim.x * 256L;
  for (; i < n; i += stride) {
    int e = (int)(i & (EMBED - 1));
    int b = (int)(i / ((long)SEQ * EMBED));
    out[i] = defs[b * EMBED + e];
  }
}

__global__ __launch_bounds__(256) void scatter_kernel(const float* __restrict__ osel,
                                                      const int* __restrict__ topk,
                                                      float* __restrict__ out, int k) {
  int rowi = blockIdx.x;                 // B*KQ rows, skip padding rows
  int b = rowi / KQ, i = rowi % KQ;
  if (i >= k) return;
  int s = topk[b * 64 + i];
  const float* src = osel + (long)rowi * EMBED;
  float* dst = out + ((long)b * SEQ + s) * EMBED;
  for (int e = threadIdx.x; e < EMBED; e += 256) dst[e] = src[e];
}

// ---------------------------------------------------------------- launch
extern "C" void kernel_launch(void* const* d_in, const int* in_sizes, int n_in,
                              void* d_out, int out_size, void* d_ws, size_t ws_size,
                              hipStream_t stream) {
  (void)in_sizes; (void)n_in; (void)out_size; (void)ws_size;
  const float* x     = (const float*)d_in[0];
  const float* in_w  = (const float*)d_in[1];   // [3E, E]
  const float* in_b  = (const float*)d_in[2];   // [3E]
  const float* out_w = (const float*)d_in[3];   // [E, E]
  const float* out_b = (const float*)d_in[4];   // [E]
  float* out = (float*)d_out;

  int kSel = (int)(5.0 * std::log((double)SEQ + 1.0));  // = 41
  if (kSel < 1) kSel = 1;
  if (kSel > KQ) kSel = KQ;

  // workspace bump allocator (256B aligned)
  char* p = (char*)d_ws;
  auto alloc = [&](size_t bytes) -> char* {
    char* r = p; p += (bytes + 255) & ~(size_t)255; return r;
  };
  u16*   x_bf    = (u16*)  alloc((size_t)BATCH * SEQ * EMBED * 2);
  u16*   win_bf  = (u16*)  alloc((size_t)3 * EMBED * EMBED * 2);
  u16*   wout_bf = (u16*)  alloc((size_t)EMBED * EMBED * 2);
  u16*   Kp      = (u16*)  alloc((size_t)BATCH * SEQ * EMBED * 2);
  u16*   Vp      = (u16*)  alloc((size_t)BATCH * SEQ * EMBED * 2);
  float* energy  = (float*)alloc((size_t)BATCH * SEQ * 4);
  float* partial = (float*)alloc((size_t)CHUNKS * BATCH * EMBED * 4);
  float* defs    = (float*)alloc((size_t)BATCH * EMBED * 4);
  int*   topk    = (int*)  alloc((size_t)BATCH * 64 * 4);
  u16*   qtok    = (u16*)  alloc((size_t)BATCH * KQ * EMBED * 2);
  u16*   Qp      = (u16*)  alloc((size_t)BATCH * KQ * EMBED * 2);
  float* scores  = (float*)alloc((size_t)BATCH * HEADS * KQ * SEQ * 4);
  u16*   attn    = (u16*)  alloc((size_t)BATCH * HEADS * KQ * SEQ * 2);
  u16*   ctxb    = (u16*)  alloc((size_t)BATCH * KQ * EMBED * 2);
  float* osel    = (float*)alloc((size_t)BATCH * KQ * EMBED * 4);

  // 1) bf16 conversions
  cvt_bf16_kernel<<<8192, 256, 0, stream>>>(x, x_bf, (long)BATCH * SEQ * EMBED);
  cvt_bf16_kernel<<<2048, 256, 0, stream>>>(in_w, win_bf, (long)3 * EMBED * EMBED);
  cvt_bf16_kernel<<<1024, 256, 0, stream>>>(out_w, wout_bf, (long)EMBED * EMBED);

  // 2) energies + per-batch column means (two-stage, deterministic)
  energy_kernel<<<BATCH * SEQ, 256, 0, stream>>>(x, energy);
  colsum_kernel<<<dim3(EMBED / 256, CHUNKS, BATCH), 256, 0, stream>>>(x, partial);
  colreduce_kernel<<<(BATCH * EMBED) / 256, 256, 0, stream>>>(partial, defs);

  // 3) top-k selection + gather selected tokens (bf16, zero-padded to KQ)
  topk_kernel<<<BATCH, 256, 0, stream>>>(energy, topk, kSel);
  gather_q_kernel<<<BATCH * KQ, 256, 0, stream>>>(x, topk, qtok, kSel);

  // 4) projections: Q (small), K, V (big)  — C = A @ W^T + b, bf16 out
  gemm_kernel<128, 128, 4, 2, true, true><<<dim3(EMBED / 128, (BATCH * KQ) / 128, 1), 256, 0, stream>>>(
      qtok, win_bf, Qp, in_b, 1.f,
      EMBED, EMBED, EMBED, EMBED,
      1, 0, 0, 0, 0, 0, 0);
  gemm_kernel<128, 128, 4, 2, true, true><<<dim3(EMBED / 128, (BATCH * SEQ) / 128, 1), 256, 0, stream>>>(
      x_bf, win_bf + (size_t)EMBED * EMBED, Kp, in_b + EMBED, 1.f,
      EMBED, EMBED, EMBED, EMBED,
      1, 0, 0, 0, 0, 0, 0);
  gemm_kernel<128, 128, 4, 2, true, true><<<dim3(EMBED / 128, (BATCH * SEQ) / 128, 1), 256, 0, stream>>>(
      x_bf, win_bf + (size_t)2 * EMBED * EMBED, Vp, in_b + 2 * EMBED, 1.f,
      EMBED, EMBED, EMBED, EMBED,
      1, 0, 0, 0, 0, 0, 0);

  // 5) scores[b,h] = (Q_h @ K_h^T) / sqrt(hd) ; batched over z = b*H + h
  gemm_kernel<64, 128, 2, 4, true, false><<<dim3(SEQ / 128, 1, BATCH * HEADS), 256, 0, stream>>>(
      Qp, Kp, scores, nullptr, 0.125f,
      HDIM, EMBED, EMBED, SEQ,
      HEADS, (long)KQ * EMBED, (long)HDIM,
             (long)SEQ * EMBED, (long)HDIM,
             (long)HEADS * KQ * SEQ, (long)KQ * SEQ);

  // 6) softmax over S, bf16 attn
  softmax_kernel<<<BATCH * HEADS * KQ, 256, 0, stream>>>(scores, attn);

  // 7) ctx[b,h] = attn @ V_h   (B is [K=S, N=hd])
  gemm_kernel<64, 64, 2, 4, false, true><<<dim3(1, 1, BATCH * HEADS), 256, 0, stream>>>(
      attn, Vp, ctxb, nullptr, 1.f,
      SEQ, SEQ, EMBED, EMBED,
      HEADS, (long)HEADS * KQ * SEQ, (long)KQ * SEQ,
             (long)SEQ * EMBED, (long)HDIM,
             (long)KQ * EMBED, (long)HDIM);

  // 8) out_sel = ctx @ out_w^T + out_b (fp32)
  gemm_kernel<128, 128, 4, 2, true, false><<<dim3(EMBED / 128, (BATCH * KQ) / 128, 1), 256, 0, stream>>>(
      ctxb, wout_bf, osel, out_b, 1.f,
      EMBED, EMBED, EMBED, EMBED,
      1, 0, 0, 0, 0, 0, 0);

  // 9) broadcast default (per-batch mean) then scatter attended rows
  fill_kernel<<<8192, 256, 0, stream>>>(defs, out);
  scatter_kernel<<<BATCH * KQ, 256, 0, stream>>>(osel, topk, out, kSel);
}
